// GATLinkPredictor_15625091023234
// MI455X (gfx1250) — matrix operand order repaired and checked by reading, not compiled, not run on hardware
//
#include <hip/hip_runtime.h>
#include <hip/hip_bf16.h>
#include <math.h>

typedef _Float16 v16h __attribute__((ext_vector_type(16)));
typedef _Float16 v8h  __attribute__((ext_vector_type(8)));
typedef float    v8f  __attribute__((ext_vector_type(8)));
typedef float    v4f  __attribute__((ext_vector_type(4)));
typedef int      v4i  __attribute__((ext_vector_type(4)));

#define HID   128
#define HEADS 4
#define DH    32

// ---------------------------------------------------------------------------
// small helpers
// ---------------------------------------------------------------------------
__device__ __forceinline__ float elu1(float x) { return x > 0.f ? x : (expf(x) - 1.f); }

__global__ void zero_f32_kernel(float* __restrict__ p, size_t n) {
    size_t i = (size_t)blockIdx.x * blockDim.x + threadIdx.x;
    if (i < n) p[i] = 0.f;
}

__global__ void conv_f16_kernel(const float* __restrict__ x, _Float16* __restrict__ xh, size_t n) {
    size_t i = (size_t)blockIdx.x * blockDim.x + threadIdx.x;
    if (i < n) xh[i] = (_Float16)x[i];
}

// w: [K][Ncol] row-major fp32  ->  wt: [Ncol][K] f16 (column-major slices, contiguous K)
__global__ void conv_transpose_f16_kernel(const float* __restrict__ w, _Float16* __restrict__ wt,
                                          int K, int Ncol) {
    int t = blockIdx.x * blockDim.x + threadIdx.x;
    if (t >= K * Ncol) return;
    int k = t / Ncol, n = t % Ncol;
    wt[(size_t)n * K + k] = (_Float16)w[(size_t)k * Ncol + n];
}

// ---------------------------------------------------------------------------
// Node GEMM: h[N,128] = xh[N,128](f16) @ W (via wt = W^T in f16), f32 accum.
// Weights staged once per workgroup into LDS (32KB of the 320KB WGP LDS);
// one wave per 16-row strip; 8 column tiles x 4 K-steps = 32 WMMAs per wave,
// B operands fed from ds_load, A from two b128 global loads per K-step.
// ---------------------------------------------------------------------------
__global__ void gemm_node_wmma(const _Float16* __restrict__ xh,
                               const _Float16* __restrict__ wt,   // [128 cols][128 K]
                               float* __restrict__ out, int N) {
    __shared__ _Float16 wlds[HID * HID];                     // 32 KB
    {
        const uint4* s4 = (const uint4*)wt;
        uint4*       d4 = (uint4*)wlds;
        for (int i = threadIdx.x; i < (HID * HID) / 8; i += blockDim.x) d4[i] = s4[i];
    }
    __syncthreads();

    int wave   = blockIdx.x * (blockDim.x >> 5) + (threadIdx.x >> 5);
    int ntiles = (N + 15) >> 4;
    if (wave >= ntiles) return;                 // wave-uniform exit (EXEC stays full)
    int lane  = threadIdx.x & 31;
    int rlane = lane & 15;
    int half  = lane >> 4;
    int arow  = wave * 16 + rlane;
    if (arow >= N) arow = N - 1;                // clamp, keeps all lanes active
    const _Float16* arowp = xh + (size_t)arow * HID;

    v8f zacc = {0.f, 0.f, 0.f, 0.f, 0.f, 0.f, 0.f, 0.f};
    v8f acc[8];
#pragma unroll
    for (int i = 0; i < 8; ++i) acc[i] = zacc;

#pragma unroll
    for (int k0 = 0; k0 < HID; k0 += 32) {
        // A k order used here (contiguous, same as B): lane half 0 -> K {0..15}, half 1 -> K {16..31}
        v8h lo = *(const v8h*)(arowp + k0 + half * 16);
        v8h hi = *(const v8h*)(arowp + k0 + half * 16 + 8);
        v16h a;
#pragma unroll
        for (int j = 0; j < 8; ++j) { a[j] = lo[j]; a[j + 8] = hi[j]; }
#pragma unroll
        for (int ct = 0; ct < 8; ++ct) {
            int col = ct * 16 + rlane;
            // B layout: lane holds column `col`, 16 contiguous K (half selects K block)
            v16h b = *(const v16h*)(wlds + (size_t)col * HID + k0 + half * 16);
            acc[ct] = __builtin_amdgcn_wmma_f32_16x16x32_f16(
                false, a, false, b, (short)0, acc[ct], false, false);
        }
    }
#pragma unroll
    for (int ct = 0; ct < 8; ++ct) {
        int col = ct * 16 + rlane;
#pragma unroll
        for (int v = 0; v < 8; ++v) {
            int row = wave * 16 + v + half * 8;
            if (row < N) out[(size_t)row * HID + col] = acc[ct][v];
        }
    }
}

// ---------------------------------------------------------------------------
// Attention projections: alpha[n][h] = sum_d h[n,h*32+d] * a[h][d]
// ---------------------------------------------------------------------------
__global__ void alphas_kernel(const float* __restrict__ h,
                              const float* __restrict__ a_src, const float* __restrict__ a_dst,
                              float* __restrict__ asrc, float* __restrict__ adst, int N) {
    int t = blockIdx.x * blockDim.x + threadIdx.x;
    if (t >= N * HEADS) return;
    int n = t >> 2, hh = t & 3;
    const float* hp = h + (size_t)n * HID + hh * DH;
    const float* as = a_src + hh * DH;
    const float* ad = a_dst + hh * DH;
    float ss = 0.f, sd = 0.f;
#pragma unroll
    for (int d = 0; d < DH; ++d) { float v = hp[d]; ss += v * as[d]; sd += v * ad[d]; }
    asrc[t] = ss;
    adst[t] = sd;
}

// ---------------------------------------------------------------------------
// Edge pass 1: e = leaky_relu(asrc[src]+adst[dst]); atomic max into m[dst]
// m is int bits of a non-negative float (init 0.0f == 0 bits); positive-float
// int ordering makes atomicMax(int) exact; reference clamps the max at 0.
// ---------------------------------------------------------------------------
__global__ void edge_pass1(const int* __restrict__ src, const int* __restrict__ dst,
                           const float* __restrict__ asrc, const float* __restrict__ adst,
                           float* __restrict__ ebuf, int* __restrict__ m, int E) {
    int e = blockIdx.x * blockDim.x + threadIdx.x;
    if (e >= E) return;
    int s = src[e], d = dst[e];
    v4f as = *(const v4f*)(asrc + (size_t)s * 4);
    v4f ad = *(const v4f*)(adst + (size_t)d * 4);
    v4f ev;
#pragma unroll
    for (int hh = 0; hh < 4; ++hh) {
        float x = as[hh] + ad[hh];
        x = (x >= 0.f) ? x : 0.2f * x;
        ev[hh] = x;
        if (x > 0.f) atomicMax(&m[(size_t)d * 4 + hh], __float_as_int(x));
    }
    *(v4f*)(ebuf + (size_t)e * 4) = ev;
}

// Edge pass 2: a = exp(e - m[dst]); atomic add into s[dst]
__global__ void edge_pass2(const int* __restrict__ dst, const int* __restrict__ m,
                           float* __restrict__ ebuf, float* __restrict__ s, int E) {
    int e = blockIdx.x * blockDim.x + threadIdx.x;
    if (e >= E) return;
    int d = dst[e];
    v4f ev = *(const v4f*)(ebuf + (size_t)e * 4);
    v4i mv = *(const v4i*)(m + (size_t)d * 4);
#pragma unroll
    for (int hh = 0; hh < 4; ++hh) {
        float a = expf(ev[hh] - __int_as_float(mv[hh]));
        ev[hh] = a;
        unsafeAtomicAdd(&s[(size_t)d * 4 + hh], a);
    }
    *(v4f*)(ebuf + (size_t)e * 4) = ev;
}

// Edge pass 3: agg[dst][:] += h[src][:] * (a / (s[dst]+1e-10)); 128 threads/edge
__global__ void edge_pass3(const int* __restrict__ src, const int* __restrict__ dst,
                           const float* __restrict__ ebuf, const float* __restrict__ s,
                           const float* __restrict__ h, float* __restrict__ agg, int E) {
    long long t = (long long)blockIdx.x * blockDim.x + threadIdx.x;
    int e = (int)(t >> 7);
    int d = (int)(t & 127);
    if (e >= E) return;
    int si = src[e], di = dst[e];
    int hh = d >> 5;
    float a  = ebuf[(size_t)e * 4 + hh];
    float ss = s[(size_t)di * 4 + hh];
    float coef = a / (ss + 1e-10f);
    float val = h[(size_t)si * HID + d] * coef;
    unsafeAtomicAdd(&agg[(size_t)di * HID + d], val);
}

// x_new = elu(agg) + x_old  (fp32 out + f16 reconvert for next GEMM)
__global__ void residual_kernel(const float* __restrict__ agg, const float* __restrict__ xold,
                                float* __restrict__ xcur, _Float16* __restrict__ xh, size_t n) {
    size_t i = (size_t)blockIdx.x * blockDim.x + threadIdx.x;
    if (i >= n) return;
    float v = elu1(agg[i]) + xold[i];
    xcur[i] = v;
    xh[i]   = (_Float16)v;
}

// ---------------------------------------------------------------------------
// Link GEMM: hid[P,128] = relu([z[src]||z[dst]] @ lp_w1 + b1), K=256, WMMA f16.
// lp_w1^T staged in LDS (64 KB); A rows gathered per-lane from zh.
// ---------------------------------------------------------------------------
__global__ void gemm_link_wmma(const _Float16* __restrict__ zh,
                               const int* __restrict__ lsrc, const int* __restrict__ ldst,
                               const _Float16* __restrict__ wt1,   // [128 cols][256 K]
                               const float* __restrict__ b1,
                               float* __restrict__ hid, int P) {
    __shared__ _Float16 wlds[2 * HID * HID];                 // 64 KB
    {
        const uint4* s4 = (const uint4*)wt1;
        uint4*       d4 = (uint4*)wlds;
        for (int i = threadIdx.x; i < (2 * HID * HID) / 8; i += blockDim.x) d4[i] = s4[i];
    }
    __syncthreads();

    int wave   = blockIdx.x * (blockDim.x >> 5) + (threadIdx.x >> 5);
    int ntiles = (P + 15) >> 4;
    if (wave >= ntiles) return;
    int lane  = threadIdx.x & 31;
    int rlane = lane & 15;
    int half  = lane >> 4;
    int pair  = wave * 16 + rlane;
    if (pair >= P) pair = P - 1;
    const _Float16* srow = zh + (size_t)lsrc[pair] * HID;
    const _Float16* drow = zh + (size_t)ldst[pair] * HID;

    v8f zacc = {0.f, 0.f, 0.f, 0.f, 0.f, 0.f, 0.f, 0.f};
    v8f acc[8];
#pragma unroll
    for (int i = 0; i < 8; ++i) acc[i] = zacc;

#pragma unroll
    for (int k0 = 0; k0 < 2 * HID; k0 += 32) {
        int s1 = k0 + half * 16;       // 16-contig chunk, never straddles the 128 boundary
        int s2 = s1 + 8;
        const _Float16* p1 = (s1 < HID) ? (srow + s1) : (drow + s1 - HID);
        const _Float16* p2 = (s2 < HID) ? (srow + s2) : (drow + s2 - HID);
        v8h lo = *(const v8h*)p1;
        v8h hi = *(const v8h*)p2;
        v16h a;
#pragma unroll
        for (int j = 0; j < 8; ++j) { a[j] = lo[j]; a[j + 8] = hi[j]; }
#pragma unroll
        for (int ct = 0; ct < 8; ++ct) {
            int col = ct * 16 + rlane;
            v16h b = *(const v16h*)(wlds + (size_t)col * (2 * HID) + k0 + half * 16);
            acc[ct] = __builtin_amdgcn_wmma_f32_16x16x32_f16(
                false, a, false, b, (short)0, acc[ct], false, false);
        }
    }
#pragma unroll
    for (int ct = 0; ct < 8; ++ct) {
        int col = ct * 16 + rlane;
        float bias = b1[col];
#pragma unroll
        for (int v = 0; v < 8; ++v) {
            int row = wave * 16 + v + half * 8;
            if (row < P) hid[(size_t)row * HID + col] = fmaxf(acc[ct][v] + bias, 0.f);
        }
    }
}

// logits[p] = hid[p,:] . w2 + b2
__global__ void logits_kernel(const float* __restrict__ hid, const float* __restrict__ w2,
                              const float* __restrict__ b2, float* __restrict__ out, int P) {
    int p = blockIdx.x * blockDim.x + threadIdx.x;
    if (p >= P) return;
    const float* hp = hid + (size_t)p * HID;
    float acc = 0.f;
#pragma unroll
    for (int c = 0; c < HID; c += 4) {
        v4f h4 = *(const v4f*)(hp + c);
        v4f w4 = *(const v4f*)(w2 + c);
        acc += h4[0] * w4[0] + h4[1] * w4[1] + h4[2] * w4[2] + h4[3] * w4[3];
    }
    out[p] = acc + b2[0];
}

// ---------------------------------------------------------------------------
// host launch
// ---------------------------------------------------------------------------
extern "C" void kernel_launch(void* const* d_in, const int* in_sizes, int n_in,
                              void* d_out, int out_size, void* d_ws, size_t ws_size,
                              hipStream_t stream) {
    const float* embed  = (const float*)d_in[0];
    const float* W1     = (const float*)d_in[1];
    const float* a_src1 = (const float*)d_in[2];
    const float* a_dst1 = (const float*)d_in[3];
    const float* W2     = (const float*)d_in[4];
    const float* a_src2 = (const float*)d_in[5];
    const float* a_dst2 = (const float*)d_in[6];
    const float* lp_w1  = (const float*)d_in[7];
    const float* lp_b1  = (const float*)d_in[8];
    const float* lp_w2  = (const float*)d_in[9];
    const float* lp_b2  = (const float*)d_in[10];
    const int*   eidx   = (const int*)d_in[11];
    const int*   lsrc   = (const int*)d_in[12];
    const int*   ldst   = (const int*)d_in[13];

    const int N = in_sizes[0] / HID;
    const int E = in_sizes[11] / 2;
    const int P = in_sizes[12];
    const int* src = eidx;
    const int* dst = eidx + E;

    // workspace layout (256B aligned slabs)
    char*  base = (char*)d_ws;
    size_t off  = 0;
    auto alloc = [&](size_t bytes) -> void* {
        off = (off + 255) & ~(size_t)255;
        void* p = base + off;
        off += bytes;
        return p;
    };
    _Float16* xh   = (_Float16*)alloc((size_t)N * HID * 2);
    float*    hbuf = (float*)   alloc((size_t)N * HID * 4);
    float*    agg  = (float*)   alloc((size_t)N * HID * 4);
    float*    xcur = (float*)   alloc((size_t)N * HID * 4);
    float*    asrc = (float*)   alloc((size_t)N * HEADS * 4);
    float*    adst = (float*)   alloc((size_t)N * HEADS * 4);
    int*      mmax = (int*)     alloc((size_t)N * HEADS * 4);
    float*    ssum = (float*)   alloc((size_t)N * HEADS * 4);
    float*    ebuf = (float*)   alloc((size_t)E * HEADS * 4);
    _Float16* wt   = (_Float16*)alloc((size_t)HID * HID * 2);
    _Float16* wt1  = (_Float16*)alloc((size_t)(2 * HID) * HID * 2);
    float*    hid  = (float*)   alloc((size_t)P * HID * 4);
    (void)ws_size;

    const int TB = 256;
    auto blks  = [&](size_t n) { return (unsigned)((n + TB - 1) / TB); };
    const size_t NH = (size_t)N * HID;

    int ntilesN = (N + 15) / 16;
    unsigned gemmN_blocks = (unsigned)((ntilesN + 7) / 8);
    int ntilesP = (P + 15) / 16;
    unsigned gemmP_blocks = (unsigned)((ntilesP + 7) / 8);
    unsigned pass3_blocks = (unsigned)(((size_t)E * HID + TB - 1) / TB);

    // initial f16 features
    conv_f16_kernel<<<blks(NH), TB, 0, stream>>>(embed, xh, NH);

    auto run_layer = [&](const float* W, const float* a_s, const float* a_d,
                         const float* xold) {
        conv_transpose_f16_kernel<<<blks((size_t)HID * HID), TB, 0, stream>>>(W, wt, HID, HID);
        gemm_node_wmma<<<gemmN_blocks, TB, 0, stream>>>(xh, wt, hbuf, N);
        alphas_kernel<<<blks((size_t)N * HEADS), TB, 0, stream>>>(hbuf, a_s, a_d, asrc, adst, N);
        zero_f32_kernel<<<blks((size_t)N * HEADS), TB, 0, stream>>>((float*)mmax, (size_t)N * HEADS);
        zero_f32_kernel<<<blks((size_t)N * HEADS), TB, 0, stream>>>(ssum, (size_t)N * HEADS);
        zero_f32_kernel<<<blks(NH), TB, 0, stream>>>(agg, NH);
        edge_pass1<<<blks(E), TB, 0, stream>>>(src, dst, asrc, adst, ebuf, mmax, E);
        edge_pass2<<<blks(E), TB, 0, stream>>>(dst, mmax, ebuf, ssum, E);
        edge_pass3<<<pass3_blocks, TB, 0, stream>>>(src, dst, ebuf, ssum, hbuf, agg, E);
        residual_kernel<<<blks(NH), TB, 0, stream>>>(agg, xold, xcur, xh, NH);
    };

    run_layer(W1, a_src1, a_dst1, embed);   // layer 1: residual against input
    run_layer(W2, a_src2, a_dst2, xcur);    // layer 2: residual against xcur (in place)

    // link predictor
    conv_transpose_f16_kernel<<<blks((size_t)(2 * HID) * HID), TB, 0, stream>>>(
        lp_w1, wt1, 2 * HID, HID);
    gemm_link_wmma<<<gemmP_blocks, TB, 0, stream>>>(xh, lsrc, ldst, wt1, lp_b1, hid, P);
    logits_kernel<<<blks(P), TB, 0, stream>>>(hid, lp_w2, lp_b2, (float*)d_out, P);
}
